// HGTEncoder_29343216566602
// MI455X (gfx1250) — matrix operand order repaired and checked by reading, not compiled, not run on hardware
//
#include <hip/hip_runtime.h>
#include <float.h>

#define B_    8
#define N_    4096
#define E_    32768
#define H_    8
#define DK_   32
#define T_    3
#define R_    6
#define DIM_  256
#define BN_   (B_ * N_)
#define RH_   (R_ * H_)   // 48

typedef __attribute__((ext_vector_type(16))) __bf16 v16bf;
typedef __attribute__((ext_vector_type(8)))  float  v8f;

// 16 bf16 values viewed either as two 16-byte vectors or a WMMA operand
union BF16x16 {
    uint4 q[2];
    v16bf v;
};

// ---------- helpers ----------

__device__ __forceinline__ unsigned short bfbits(float f) {
    union { float f; unsigned u; } v; v.f = f;
    unsigned r = v.u + 0x7FFFu + ((v.u >> 16) & 1u);   // round to nearest even
    return (unsigned short)(r >> 16);
}
// float atomic max via signed-max / unsigned-min trick (works for global & LDS,
// requires init to -FLT_MAX)
__device__ __forceinline__ void atomicMaxFloat(float* addr, float value) {
    if (value >= 0.0f) atomicMax((int*)addr, __float_as_int(value));
    else               atomicMin((unsigned int*)addr, __float_as_uint(value));
}
// load 32 consecutive bf16 (64B aligned) into floats
__device__ __forceinline__ void load32bf(const unsigned short* p, float* out) {
    const uint4* q = (const uint4*)p;
#pragma unroll
    for (int i = 0; i < 4; ++i) {
        uint4 v = q[i];
        unsigned w0 = v.x, w1 = v.y, w2 = v.z, w3 = v.w;
        out[i*8+0] = __uint_as_float((w0 & 0xFFFFu) << 16);
        out[i*8+1] = __uint_as_float(w0 & 0xFFFF0000u);
        out[i*8+2] = __uint_as_float((w1 & 0xFFFFu) << 16);
        out[i*8+3] = __uint_as_float(w1 & 0xFFFF0000u);
        out[i*8+4] = __uint_as_float((w2 & 0xFFFFu) << 16);
        out[i*8+5] = __uint_as_float(w2 & 0xFFFF0000u);
        out[i*8+6] = __uint_as_float((w3 & 0xFFFFu) << 16);
        out[i*8+7] = __uint_as_float(w3 & 0xFFFF0000u);
    }
}

// ---------- kernel 0: zero / init scratch ----------

__global__ __launch_bounds__(256) void init_kernel(float* __restrict__ accum,
                                                   float* __restrict__ smax,
                                                   float* __restrict__ denom) {
    size_t idx = (size_t)blockIdx.x * 256 + threadIdx.x;
    if (idx < (size_t)BN_ * DIM_) accum[idx] = 0.0f;
    if (idx < (size_t)B_ * RH_) { smax[idx] = -FLT_MAX; denom[idx] = 0.0f; }
}

// ---------- kernel 0b: weight convert + transpose (f32 row-major -> bf16 col-major) ----
// Wt[t][n][k] = bf16( W[t][k][n] )  so a WMMA B-fragment is 32 contiguous bytes/lane.

__global__ __launch_bounds__(256) void wconv_kernel(const float* __restrict__ W,
                                                    unsigned short* __restrict__ Wt,
                                                    int mats) {
    int idx = blockIdx.x * 256 + threadIdx.x;
    int t   = idx >> 16;                 // / (256*256)
    if (t >= mats) return;
    int rem = idx & 0xFFFF;
    int k   = rem >> 8;
    int n   = rem & 255;
    Wt[((size_t)t * DIM_ + n) * DIM_ + k] = bfbits(W[idx]);
}

// ---------- kernel 1: typed projection  x @ W[type] + b[type]  -> bf16 ----------
// grid.x = BN/16 node tiles, 512 threads = 16 waves, wave w owns columns [16w,16w+16)

__global__ __launch_bounds__(512) void proj_kernel(const float* __restrict__ x,
                                                   const int*   __restrict__ ntypes,
                                                   const unsigned short* __restrict__ Wc, // (T,256,256) bf16 col-major
                                                   const float* __restrict__ bias,        // (T,256)
                                                   unsigned short* __restrict__ out) {
    __shared__ __align__(16) unsigned short Abf[16][DIM_ + 8];
    __shared__ int ntS[16];
    const int tid   = threadIdx.x;
    const int node0 = blockIdx.x * 16;

#pragma unroll
    for (int i = 0; i < 8; ++i) {
        int idx = tid + i * 512;
        int row = idx >> 8, col = idx & 255;
        Abf[row][col] = bfbits(x[(size_t)(node0 + row) * DIM_ + col]);
    }
    if (tid < 16) ntS[tid] = ntypes[node0 + tid];
    __syncthreads();

    const int lane = tid & 31;
    const int c0   = (tid >> 5) * 16;
    const int arow = lane & 15;
    const int kq   = (lane & 16) ? 16 : 0;    // A lane-half K offset
    const int bcol = lane & 15;
    const int brow = (lane & 16) ? 16 : 0;    // B lane-half K offset

    v8f acc0 = {}, acc1 = {}, acc2 = {};
    for (int kb = 0; kb < DIM_; kb += 32) {
        BF16x16 a;
        a.q[0] = *(const uint4*)&Abf[arow][kb + kq];        // k = kb+kq .. +8
        a.q[1] = *(const uint4*)&Abf[arow][kb + kq + 8];    // k = kb+kq+8 .. +8
#pragma unroll
        for (int t = 0; t < T_; ++t) {
            BF16x16 bm;
            const unsigned short* wp =
                Wc + ((size_t)(t * DIM_ + c0 + bcol) * DIM_ + kb + brow);
            bm.q[0] = *(const uint4*)wp;          // k = kb+brow .. +8
            bm.q[1] = *(const uint4*)(wp + 8);    // k = kb+brow+8 .. +8
            v8f& acc = (t == 0) ? acc0 : (t == 1) ? acc1 : acc2;
            acc = __builtin_amdgcn_wmma_f32_16x16x32_bf16(
                    false, a.v, false, bm.v, (short)0, acc, false, false);
        }
    }

#pragma unroll
    for (int i = 0; i < 8; ++i) {
        int row = ((lane >> 4) & 1) * 8 + i;      // C/D layout: lanes 16-31 hold M+8
        int col = c0 + (lane & 15);
        int t = ntS[row];
        float v = (t == 0) ? acc0[i] : (t == 1) ? acc1[i] : acc2[i];
        v += bias[t * DIM_ + col];
        out[(size_t)(node0 + row) * DIM_ + col] = bfbits(v);
    }
}

// ---------- kernel 2: edge scores + hierarchical max ----------
// thread = (b,e,h); block = 256 threads (32 edges x 8 heads, single batch)

__global__ __launch_bounds__(256) void score_kernel(const int* __restrict__ eidx,
                                                    const int* __restrict__ etyp,
                                                    const int* __restrict__ emask,
                                                    const unsigned short* __restrict__ Kb,
                                                    const unsigned short* __restrict__ Qb,
                                                    const float* __restrict__ Watt,   // (R,32,32)
                                                    const float* __restrict__ relpri, // (R,H)
                                                    float* __restrict__ sbuf,
                                                    float* __restrict__ smax) {
    __shared__ float lmax[RH_];
    const int tid = threadIdx.x;
    if (tid < RH_) lmax[tid] = -FLT_MAX;
    __syncthreads();

    const int idx = blockIdx.x * 256 + tid;
    const int b   = idx / (E_ * H_);
    const int rm  = idx - b * (E_ * H_);
    const int e   = rm >> 3;
    const int h   = rm & 7;
    const int m   = emask[b * E_ + e];
    const int r   = etyp[b * E_ + e];

    float s = -FLT_MAX;
    if (m) {
        const int src = eidx[(size_t)b * 2 * E_ + e];
        const int tgt = eidx[(size_t)b * 2 * E_ + E_ + e];
        float kv[32];
        load32bf(Kb + ((size_t)(b * N_ + src) * DIM_ + h * DK_), kv);
        float u[32];
#pragma unroll
        for (int j = 0; j < 32; ++j) u[j] = 0.0f;
#pragma unroll 4
        for (int d = 0; d < DK_; ++d) {
            float kd = kv[d];
            const float4* wr = (const float4*)(Watt + ((size_t)r * DK_ + d) * DK_);
#pragma unroll
            for (int jq = 0; jq < 8; ++jq) {
                float4 w = wr[jq];
                u[jq*4+0] += kd * w.x;  u[jq*4+1] += kd * w.y;
                u[jq*4+2] += kd * w.z;  u[jq*4+3] += kd * w.w;
            }
        }
        float qv[32];
        load32bf(Qb + ((size_t)(b * N_ + tgt) * DIM_ + h * DK_), qv);
        s = 0.0f;
#pragma unroll
        for (int j = 0; j < 32; ++j) s += qv[j] * u[j];
        s *= relpri[r * H_ + h] * 0.17677669529663687f;  // 1/sqrt(32)
        atomicMaxFloat(&lmax[r * H_ + h], s);
    }
    sbuf[idx] = s;
    __syncthreads();
    if (tid < RH_) {
        float v = lmax[tid];
        if (v > -FLT_MAX) atomicMaxFloat(&smax[b * RH_ + tid], v);
    }
}

// ---------- kernel 3: exp + hierarchical denominator ----------
// 4 items/thread (same edge, 4 heads); block covers 128 edges of one batch

__global__ __launch_bounds__(256) void expsum_kernel(const int* __restrict__ etyp,
                                                     const int* __restrict__ emask,
                                                     float* __restrict__ sbuf,
                                                     const float* __restrict__ smax,
                                                     float* __restrict__ denom) {
    __shared__ float part[RH_];
    const int tid = threadIdx.x;
    if (tid < RH_) part[tid] = 0.0f;
    __syncthreads();

    const int idx0 = blockIdx.x * 1024 + tid * 4;
    const int b = idx0 / (E_ * H_);
    const int rm = idx0 - b * (E_ * H_);
    const int e = rm >> 3;
    const int h0 = rm & 7;           // 0 or 4
    const int m = emask[b * E_ + e];
    const int r = etyp[b * E_ + e];

#pragma unroll
    for (int t = 0; t < 4; ++t) {
        int h = h0 + t;
        float ex = 0.0f;
        if (m) {
            float s = sbuf[idx0 + t];
            ex = __expf(s - smax[b * RH_ + r * H_ + h]);
            atomicAdd(&part[r * H_ + h], ex);
        }
        sbuf[idx0 + t] = ex;
    }
    __syncthreads();
    if (tid < RH_) {
        float v = part[tid];
        if (v != 0.0f) atomicAdd(&denom[b * RH_ + tid], v);
    }
}

// ---------- kernel 4: messages + scatter-add ----------

__global__ __launch_bounds__(256) void message_kernel(const int* __restrict__ eidx,
                                                      const int* __restrict__ etyp,
                                                      const unsigned short* __restrict__ Vb,
                                                      const float* __restrict__ Wmsg,  // (R,32,32)
                                                      const float* __restrict__ sbuf,  // ex
                                                      const float* __restrict__ denom,
                                                      float* __restrict__ accum) {
    const int idx = blockIdx.x * 256 + threadIdx.x;
    const int b   = idx / (E_ * H_);
    const int rm  = idx - b * (E_ * H_);
    const int e   = rm >> 3;
    const int h   = rm & 7;

    float ex = sbuf[idx];
    if (ex <= 0.0f) return;
    const int r = etyp[b * E_ + e];
    float den = denom[b * RH_ + r * H_ + h];
    if (den <= 0.0f) return;
    float p = ex / fmaxf(den, 1e-30f);

    const int src = eidx[(size_t)b * 2 * E_ + e];
    const int tgt = eidx[(size_t)b * 2 * E_ + E_ + e];
    float vv[32];
    load32bf(Vb + ((size_t)(b * N_ + src) * DIM_ + h * DK_), vv);
    float u[32];
#pragma unroll
    for (int j = 0; j < 32; ++j) u[j] = 0.0f;
#pragma unroll 4
    for (int d = 0; d < DK_; ++d) {
        float vd = vv[d];
        const float4* wr = (const float4*)(Wmsg + ((size_t)r * DK_ + d) * DK_);
#pragma unroll
        for (int jq = 0; jq < 8; ++jq) {
            float4 w = wr[jq];
            u[jq*4+0] += vd * w.x;  u[jq*4+1] += vd * w.y;
            u[jq*4+2] += vd * w.z;  u[jq*4+3] += vd * w.w;
        }
    }
    float* dst = accum + (size_t)(b * N_ + tgt) * DIM_ + h * DK_;
#pragma unroll
    for (int j = 0; j < 32; ++j) atomicAdd(dst + j, p * u[j]);
}

// ---------- kernel 5: output projection (WMMA) + bias + residual ----------

__global__ __launch_bounds__(512) void outproj_kernel(const float* __restrict__ accum,
                                                      const float* __restrict__ x,
                                                      const unsigned short* __restrict__ Woc, // (256,256) bf16 col-major
                                                      const float* __restrict__ bout,
                                                      float* __restrict__ out) {
    __shared__ __align__(16) unsigned short Abf[16][DIM_ + 8];
    const int tid   = threadIdx.x;
    const int node0 = blockIdx.x * 16;

#pragma unroll
    for (int i = 0; i < 8; ++i) {
        int idx = tid + i * 512;
        int row = idx >> 8, col = idx & 255;
        Abf[row][col] = bfbits(accum[(size_t)(node0 + row) * DIM_ + col]);
    }
    __syncthreads();

    const int lane = tid & 31;
    const int c0   = (tid >> 5) * 16;
    const int arow = lane & 15;
    const int kq   = (lane & 16) ? 16 : 0;
    const int bcol = lane & 15;
    const int brow = (lane & 16) ? 16 : 0;

    v8f acc = {};
    for (int kb = 0; kb < DIM_; kb += 32) {
        BF16x16 a, bm;
        a.q[0] = *(const uint4*)&Abf[arow][kb + kq];
        a.q[1] = *(const uint4*)&Abf[arow][kb + kq + 8];
        const unsigned short* wp =
            Woc + ((size_t)(c0 + bcol) * DIM_ + kb + brow);
        bm.q[0] = *(const uint4*)wp;
        bm.q[1] = *(const uint4*)(wp + 8);
        acc = __builtin_amdgcn_wmma_f32_16x16x32_bf16(
                false, a.v, false, bm.v, (short)0, acc, false, false);
    }

#pragma unroll
    for (int i = 0; i < 8; ++i) {
        int row = ((lane >> 4) & 1) * 8 + i;
        int col = c0 + (lane & 15);
        size_t o = (size_t)(node0 + row) * DIM_ + col;
        out[o] = acc[i] + bout[col] + x[o];   // residual before LN
    }
}

// ---------- kernel 6: layernorm + node mask (in place, 1 wave / row) ----------

__global__ __launch_bounds__(256) void ln_kernel(float* __restrict__ out,
                                                 const float* __restrict__ gamma,
                                                 const float* __restrict__ beta,
                                                 const int* __restrict__ nmask) {
    const int row  = blockIdx.x * 8 + (threadIdx.x >> 5);
    const int lane = threadIdx.x & 31;
    float* p = out + (size_t)row * DIM_;
    float v[8];
#pragma unroll
    for (int i = 0; i < 8; ++i) v[i] = p[lane * 8 + i];
    float s = 0.0f;
#pragma unroll
    for (int i = 0; i < 8; ++i) s += v[i];
#pragma unroll
    for (int off = 16; off > 0; off >>= 1) s += __shfl_xor(s, off, 32);
    float mu = s * (1.0f / 256.0f);
    float vs = 0.0f;
#pragma unroll
    for (int i = 0; i < 8; ++i) { float d = v[i] - mu; vs += d * d; }
#pragma unroll
    for (int off = 16; off > 0; off >>= 1) vs += __shfl_xor(vs, off, 32);
    float inv = rsqrtf(vs * (1.0f / 256.0f) + 1e-5f);
    float m = (float)nmask[row];
#pragma unroll
    for (int i = 0; i < 8; ++i) {
        int c = lane * 8 + i;
        p[c] = ((v[i] - mu) * inv * gamma[c] + beta[c]) * m;
    }
}

// ---------- launch ----------

extern "C" void kernel_launch(void* const* d_in, const int* in_sizes, int n_in,
                              void* d_out, int out_size, void* d_ws, size_t ws_size,
                              hipStream_t stream) {
    (void)in_sizes; (void)n_in; (void)out_size; (void)ws_size;
    const float* x      = (const float*)d_in[0];
    const int*   eidx   = (const int*)d_in[1];
    const int*   ntyp   = (const int*)d_in[2];
    const int*   etyp   = (const int*)d_in[3];
    const int*   nmask  = (const int*)d_in[4];
    const int*   emask  = (const int*)d_in[5];
    const float* Wk     = (const float*)d_in[6];
    const float* bk     = (const float*)d_in[7];
    const float* Wq     = (const float*)d_in[8];
    const float* bq     = (const float*)d_in[9];
    const float* Wv     = (const float*)d_in[10];
    const float* bv     = (const float*)d_in[11];
    const float* relpri = (const float*)d_in[12];
    const float* Watt   = (const float*)d_in[13];
    const float* Wmsg   = (const float*)d_in[14];
    const float* Wout   = (const float*)d_in[15];
    const float* bout   = (const float*)d_in[16];
    const float* gamma  = (const float*)d_in[17];
    const float* beta   = (const float*)d_in[18];
    float* out = (float*)d_out;

    char* ws = (char*)d_ws;
    size_t off = 0;
    auto take = [&](size_t bytes) -> char* {
        char* p = ws + off;
        off += (bytes + 255) & ~(size_t)255;
        return p;
    };
    unsigned short* Kb  = (unsigned short*)take((size_t)BN_ * DIM_ * 2);
    unsigned short* Qb  = (unsigned short*)take((size_t)BN_ * DIM_ * 2);
    unsigned short* Vb  = (unsigned short*)take((size_t)BN_ * DIM_ * 2);
    float* sbuf  = (float*)take((size_t)B_ * E_ * H_ * 4);
    float* smax  = (float*)take((size_t)B_ * RH_ * 4);
    float* denom = (float*)take((size_t)B_ * RH_ * 4);
    float* accum = (float*)take((size_t)BN_ * DIM_ * 4);
    unsigned short* Wkc = (unsigned short*)take((size_t)T_ * DIM_ * DIM_ * 2);
    unsigned short* Wqc = (unsigned short*)take((size_t)T_ * DIM_ * DIM_ * 2);
    unsigned short* Wvc = (unsigned short*)take((size_t)T_ * DIM_ * DIM_ * 2);
    unsigned short* Woc = (unsigned short*)take((size_t)DIM_ * DIM_ * 2);

    init_kernel<<<(BN_ * DIM_) / 256, 256, 0, stream>>>(accum, smax, denom);
    wconv_kernel<<<(T_ * DIM_ * DIM_) / 256, 256, 0, stream>>>(Wk, Wkc, T_);
    wconv_kernel<<<(T_ * DIM_ * DIM_) / 256, 256, 0, stream>>>(Wq, Wqc, T_);
    wconv_kernel<<<(T_ * DIM_ * DIM_) / 256, 256, 0, stream>>>(Wv, Wvc, T_);
    wconv_kernel<<<(DIM_ * DIM_) / 256, 256, 0, stream>>>(Wout, Woc, 1);
    proj_kernel<<<BN_ / 16, 512, 0, stream>>>(x, ntyp, Wkc, bk, Kb);
    proj_kernel<<<BN_ / 16, 512, 0, stream>>>(x, ntyp, Wqc, bq, Qb);
    proj_kernel<<<BN_ / 16, 512, 0, stream>>>(x, ntyp, Wvc, bv, Vb);
    score_kernel<<<(B_ * E_ * H_) / 256, 256, 0, stream>>>(eidx, etyp, emask, Kb, Qb,
                                                           Watt, relpri, sbuf, smax);
    expsum_kernel<<<(B_ * E_ * H_) / 1024, 256, 0, stream>>>(etyp, emask, sbuf, smax, denom);
    message_kernel<<<(B_ * E_ * H_) / 256, 256, 0, stream>>>(eidx, etyp, Vb, Wmsg,
                                                             sbuf, denom, accum);
    outproj_kernel<<<BN_ / 16, 512, 0, stream>>>(accum, x, Woc, bout, out);
    ln_kernel<<<BN_ / 8, 256, 0, stream>>>(out, gamma, beta, nmask);
}
